// Attention_88785563943675
// MI455X (gfx1250) — matrix operand order repaired and checked by reading, not compiled, not run on hardware
//
#include <hip/hip_runtime.h>
#include <hip/hip_bf16.h>

typedef __attribute__((ext_vector_type(16))) _Float16 v16h;
typedef __attribute__((ext_vector_type(8)))  _Float16 v8h;
typedef __attribute__((ext_vector_type(8)))  float    v8f;
typedef __attribute__((ext_vector_type(4)))  unsigned int v4u;
typedef __attribute__((ext_vector_type(8)))  int      v8i;
typedef __attribute__((ext_vector_type(4)))  int      v4i;

#define NB_HEAD 4
#define HDIM    50
#define DMODEL  200
#define BATCH   4
#define SEQ     4096
#define DPAD    64            // head dim padded 50 -> 64 (2 WMMA K-steps)
#define NPAD    208           // 13 * 16 output tiles, real N = 200
#define ROWS    (BATCH*SEQ)   // 16384
#define NBLK    (SEQ/32)      // 128 key blocks

__device__ __forceinline__ v8h  ld8h (const _Float16* p){ return *reinterpret_cast<const v8h*>(p); }
__device__ __forceinline__ v16h ld16h(const _Float16* p){ return *reinterpret_cast<const v16h*>(p); }
__device__ __forceinline__ v16h cat8(v8h lo, v8h hi){
  return __builtin_shufflevector(lo, hi, 0,1,2,3,4,5,6,7,8,9,10,11,12,13,14,15);
}

// pure-VALU butterfly within each 16-lane row: DPP16 ROW_XMASK (0x160|mask)
template<int CTRL>
__device__ __forceinline__ float dpp_xor(float v){
  int r = __builtin_amdgcn_update_dpp(0, __builtin_bit_cast(int, v), CTRL, 0xF, 0xF, true);
  return __builtin_bit_cast(float, r);
}
__device__ __forceinline__ float redmax16(float v){
  v = fmaxf(v, dpp_xor<0x161>(v));
  v = fmaxf(v, dpp_xor<0x162>(v));
  v = fmaxf(v, dpp_xor<0x164>(v));
  v = fmaxf(v, dpp_xor<0x168>(v));
  return v;
}
__device__ __forceinline__ float redsum16(float v){
  v += dpp_xor<0x161>(v);
  v += dpp_xor<0x162>(v);
  v += dpp_xor<0x164>(v);
  v += dpp_xor<0x168>(v);
  return v;
}

// ---- TDM: 2-D tile (tile_d0 contiguous elements x tile_d1 lines) -> LDS ----
// D# per CDNA5 ISA; data_size = 2 bytes (f16).
// This toolchain exposes the 6-arg builtin: (g0, g1, g2, g3, g4, cpol).
__device__ __forceinline__ void tdm_load_2d_f16(unsigned lds_addr, const void* gaddr,
                                                unsigned tile_d0, unsigned tile_d1,
                                                unsigned long long stride0)
{
  unsigned long long ga = (unsigned long long)(uintptr_t)gaddr;
  v4u g0;
  g0[0] = 1u;                                            // count=1, user descriptor
  g0[1] = lds_addr;                                      // lds_addr[31:0]
  g0[2] = (unsigned)(ga & 0xFFFFFFFFu);                  // global_addr[31:0]
  g0[3] = (unsigned)((ga >> 32) & 0x1FFFFFFu) | (2u << 30); // addr[56:32] | type=2
  v8i g1;
  g1[0] = (int)(1u << 16);                               // wg_mask=0, data_size=1 (2B)
  g1[1] = (int)((tile_d0 & 0xFFFFu) << 16);              // tensor_dim0[15:0]
  g1[2] = (int)(((tile_d0 >> 16) & 0xFFFFu) | ((tile_d1 & 0xFFFFu) << 16)); // dim0 hi | tensor_dim1 lo
  g1[3] = (int)(((tile_d1 >> 16) & 0xFFFFu) | ((tile_d0 & 0xFFFFu) << 16)); // dim1 hi | tile_dim0
  g1[4] = (int)(tile_d1 & 0xFFFFu);                      // tile_dim1 | tile_dim2=0
  g1[5] = (int)(stride0 & 0xFFFFFFFFu);                  // tensor_dim0_stride lo32
  g1[6] = (int)((stride0 >> 32) & 0xFFFFu);              // stride0 hi16 | stride1 lo16=0
  g1[7] = 0;
  v4i z4 = {0, 0, 0, 0};
  v8i z8 = {0, 0, 0, 0, 0, 0, 0, 0};
  __builtin_amdgcn_tensor_load_to_lds(g0, g1, z4, z4, z8, 0);
}

// ---------------- projections: fp32 VALU dot products, f16 padded output ----
__global__ __launch_bounds__(256) void proj_kernel(const float* __restrict__ X,
                                                   const float* __restrict__ W,
                                                   _Float16* __restrict__ Y)
{
  const int idx = blockIdx.x * 256 + threadIdx.x;   // over ROWS*DPAD
  const int row = idx >> 6;
  const int e   = idx & 63;
  float a = 0.f;
  if (e < HDIM){
    const float* xr = X + (size_t)row * DMODEL;
    #pragma unroll 4
    for (int k = 0; k < DMODEL; ++k) a = fmaf(xr[k], W[k*HDIM + e], a);
  }
  Y[idx] = (_Float16)a;                              // zero pad e>=50
}

// V projection writing TRANSPOSED f16 output: YT[b][e][s]  (TDM-friendly tiles)
__global__ __launch_bounds__(256) void projT_kernel(const float* __restrict__ X,
                                                    const float* __restrict__ W,
                                                    _Float16* __restrict__ YT)
{
  const int idx = blockIdx.x * 256 + threadIdx.x;   // over ROWS*DPAD
  const int row = idx >> 6;
  const int e   = idx & 63;
  float a = 0.f;
  if (e < HDIM){
    const float* xr = X + (size_t)row * DMODEL;
    #pragma unroll 4
    for (int k = 0; k < DMODEL; ++k) a = fmaf(xr[k], W[k*HDIM + e], a);
  }
  const int b = row >> 12;           // row / SEQ
  const int s = row & (SEQ - 1);
  YT[((size_t)b * DPAD + e) * SEQ + s] = (_Float16)a;
}

// ---- fold the 4 identical heads: WsumT[n][e] = sum_h WO[h*50+e][n], f16 ----
__global__ __launch_bounds__(256) void wsum_kernel(const float* __restrict__ WO,
                                                   _Float16* __restrict__ WsumT)
{
  const int idx = blockIdx.x * 256 + threadIdx.x;   // NPAD*DPAD = 13312
  const int n = idx >> 6;
  const int e = idx & 63;
  float s = 0.f;
  if (n < DMODEL && e < HDIM){
    #pragma unroll
    for (int h = 0; h < NB_HEAD; ++h) s += WO[(h*HDIM + e)*DMODEL + n];
  }
  WsumT[idx] = (_Float16)s;
}

// ---------------- flash attention core: wave32 WMMA f16 -> f32 --------------
// Double-buffered TDM pipeline: tile kb+1 DMA overlaps compute of tile kb.
__global__ __launch_bounds__(128) void flash_kernel(const _Float16* __restrict__ Qp,
                                                    const _Float16* __restrict__ Kp,
                                                    const _Float16* __restrict__ VpT,
                                                    _Float16* __restrict__ O16)
{
  __shared__ __attribute__((aligned(32))) _Float16 Klds[2][32][DPAD];   // keys row-major
  __shared__ __attribute__((aligned(32))) _Float16 Vtlds[2][DPAD][32];  // V transposed
  __shared__ __attribute__((aligned(32))) _Float16 Plds[4][16][32];     // per-wave P scratch

  const int tid  = threadIdx.x;       // 0..127, 4 waves
  const int lane = tid & 31;
  const int wv   = tid >> 5;
  const int col  = lane & 15;
  const int kh   = lane >> 4;

  const int b  = blockIdx.x >> 6;           // SEQ/64 = 64 q-blocks per batch
  const int q0 = (blockIdx.x & 63) * 64;
  const size_t bS = (size_t)b * SEQ;
  const _Float16* vbase = VpT + (size_t)b * DPAD * SEQ;

  // Q fragments for this wave's 16 rows, resident for the whole kernel
  v16h aq[2];
  {
    const _Float16* qrow = Qp + (bS + q0 + wv*16 + col) * DPAD;
    #pragma unroll
    for (int kc = 0; kc < 2; ++kc)
      aq[kc] = ld16h(qrow + kc*32 + kh*16);
  }

  float m_s[8], l_s[8];
  v8f acc[4];
  {
    v8f z = {};
    #pragma unroll
    for (int t = 0; t < 4; ++t) acc[t] = z;
  }
  #pragma unroll
  for (int r = 0; r < 8; ++r){ m_s[r] = -1e30f; l_s[r] = 0.f; }

  const float scale = 0.14142135623f;       // 1/sqrt(50)

  // issue tile `n` (2 TDM loads) into LDS buffer n&1
  auto issue_tile = [&](int n){
    const int s = n & 1;
    // K tile: 32 key-lines of 64 contiguous halves, line stride 64
    tdm_load_2d_f16((unsigned)(uintptr_t)&Klds[s][0][0],
                    Kp + (bS + n*32) * DPAD, DPAD, 32, DPAD);
    // V tile (pre-transposed in global): 64 d-lines of 32 contiguous keys, stride SEQ
    tdm_load_2d_f16((unsigned)(uintptr_t)&Vtlds[s][0][0],
                    vbase + n*32, 32, DPAD, SEQ);
  };

  if (wv == 0) issue_tile(0);               // prologue: tile 0 in flight

  for (int kb = 0; kb < NBLK; ++kb){
    __syncthreads();      // everyone done with the buffer we are about to refill
    if (wv == 0){
      if (kb + 1 < NBLK){
        issue_tile(kb + 1);                           // overlap with tile kb compute
        __builtin_amdgcn_s_wait_tensorcnt((short)2);  // in-order: tile kb landed
      } else {
        __builtin_amdgcn_s_wait_tensorcnt((short)0);  // last tile
      }
    }
    __syncthreads();      // tile kb visible to all waves
    const int sel = kb & 1;

    // S-tile (16 rows x 32 keys) = Q Kt : 4 WMMA
    v8f c0 = {}, c1 = {};
    #pragma unroll
    for (int kc = 0; kc < 2; ++kc){
      v16h bk0 = ld16h(&Klds[sel][col     ][kc*32 + kh*16]);   // B-layout: lane=key col
      v16h bk1 = ld16h(&Klds[sel][16 + col][kc*32 + kh*16]);
      c0 = __builtin_amdgcn_wmma_f32_16x16x32_f16(false, aq[kc], false, bk0, (short)0, c0, false, false);
      c1 = __builtin_amdgcn_wmma_f32_16x16x32_f16(false, aq[kc], false, bk1, (short)0, c1, false, false);
    }

    // online softmax (C layout: lane half <-> row group, VGPR r <-> row)
    float fac[8];
    #pragma unroll
    for (int r = 0; r < 8; ++r){
      float s0 = c0[r] * scale, s1 = c1[r] * scale;
      float mx = redmax16(fmaxf(s0, s1));
      float mn = fmaxf(m_s[r], mx);
      fac[r] = __expf(m_s[r] - mn);
      m_s[r] = mn;
      s0 = __expf(s0 - mn);  s1 = __expf(s1 - mn);
      c0[r] = s0;  c1[r] = s1;
      l_s[r] = l_s[r] * fac[r] + redsum16(s0 + s1);
    }
    #pragma unroll
    for (int t = 0; t < 4; ++t){
      #pragma unroll
      for (int r = 0; r < 8; ++r) acc[t][r] *= fac[r];
    }

    // C-layout -> A-layout for P via this wave's private LDS scratch
    #pragma unroll
    for (int r = 0; r < 8; ++r){
      const int row = r + 8*kh;
      Plds[wv][row][col]      = (_Float16)c0[r];
      Plds[wv][row][16 + col] = (_Float16)c1[r];
    }
    asm volatile("s_wait_dscnt 0" ::: "memory");   // same-wave DS RAW ordering
    v16h pa = ld16h(&Plds[wv][col][kh*16]);

    // O += P V : 4 WMMA over the 64 padded head dims
    #pragma unroll
    for (int t = 0; t < 4; ++t){
      v16h bv = ld16h(&Vtlds[sel][t*16 + col][kh*16]);
      acc[t] = __builtin_amdgcn_wmma_f32_16x16x32_f16(false, pa, false, bv, (short)0, acc[t], false, false);
    }
  }

  // normalize and emit f16 [row][64]
  #pragma unroll
  for (int r = 0; r < 8; ++r){
    const float inv = 1.0f / l_s[r];
    const size_t row = bS + q0 + wv*16 + r + 8*kh;
    #pragma unroll
    for (int t = 0; t < 4; ++t)
      O16[row*DPAD + t*16 + col] = (_Float16)(acc[t][r] * inv);
  }
}

// -------- output projection: [16384x64] @ WsumT^T -> fp32 [16384x200] -------
__global__ __launch_bounds__(256) void outproj_kernel(const _Float16* __restrict__ O16,
                                                      const _Float16* __restrict__ WsumT,
                                                      float* __restrict__ out)
{
  const int lane = threadIdx.x & 31;
  const int wv   = threadIdx.x >> 5;
  const int wave = blockIdx.x * 8 + wv;    // 0 .. 1024*13-1
  const int tm   = wave / 13;
  const int tn   = wave % 13;
  const int col  = lane & 15;
  const int kh   = lane >> 4;

  v8f acc = {};
  const _Float16* arow = O16   + (size_t)(tm*16 + col) * DPAD;
  const _Float16* brow = WsumT + (size_t)(tn*16 + col) * DPAD;
  #pragma unroll
  for (int ks = 0; ks < 2; ++ks){
    v16h a  = ld16h(arow + ks*32 + kh*16);
    v16h bm = ld16h(brow + ks*32 + kh*16);
    acc = __builtin_amdgcn_wmma_f32_16x16x32_f16(false, a, false, bm, (short)0, acc, false, false);
  }
  #pragma unroll
  for (int r = 0; r < 8; ++r){
    const int row = tm*16 + r + 8*kh;
    const int n   = tn*16 + col;
    if (n < DMODEL) out[(size_t)row*DMODEL + n] = acc[r];
  }
}

extern "C" void kernel_launch(void* const* d_in, const int* in_sizes, int n_in,
                              void* d_out, int out_size, void* d_ws, size_t ws_size,
                              hipStream_t stream)
{
  const float* q  = (const float*)d_in[0];
  const float* k  = (const float*)d_in[1];
  const float* v  = (const float*)d_in[2];
  const float* WQ = (const float*)d_in[3];
  const float* WK = (const float*)d_in[4];
  const float* WV = (const float*)d_in[5];
  const float* WO = (const float*)d_in[6];
  float* out = (float*)d_out;

  _Float16* ws = (_Float16*)d_ws;
  const size_t per = (size_t)ROWS * DPAD;   // 1,048,576 halves each
  _Float16* Qp    = ws;
  _Float16* Kp    = ws + per;
  _Float16* VpT   = ws + 2*per;             // transposed: [B][DPAD][SEQ]
  _Float16* O16   = ws + 3*per;
  _Float16* WsumT = ws + 4*per;             // 13,312 halves  (total ~8.4 MB)

  const int projBlocks = (ROWS * DPAD) / 256;          // 4096
  proj_kernel <<<projBlocks, 256, 0, stream>>>(q, WQ, Qp);
  proj_kernel <<<projBlocks, 256, 0, stream>>>(k, WK, Kp);
  projT_kernel<<<projBlocks, 256, 0, stream>>>(v, WV, VpT);
  wsum_kernel<<<(NPAD*DPAD)/256, 256, 0, stream>>>(WO, WsumT);
  flash_kernel<<<BATCH * (SEQ/64), 128, 0, stream>>>(Qp, Kp, VpT, O16);
  outproj_kernel<<<(ROWS/16) * 13 / 8, 256, 0, stream>>>(O16, WsumT, out);
}
